// BiLSTM_CRF_32847909880581
// MI455X (gfx1250) — matrix operand order repaired and checked by reading, not compiled, not run on hardware
//
#include <hip/hip_runtime.h>
#include <hip/hip_bf16.h>
#include <math.h>

// ---------------------------------------------------------------------------
// BiLSTM-CRF forward for MI455X (gfx1250, wave32, WMMA 16x16x32 bf16).
// ---------------------------------------------------------------------------

typedef __bf16 bf16;
typedef __bf16 v16bf __attribute__((ext_vector_type(16)));
typedef float  v8f   __attribute__((ext_vector_type(8)));
typedef float  f32x4 __attribute__((ext_vector_type(4)));

union FragBF { v16bf v; f32x4 q[2]; };

// Problem constants
#define BN    64
#define TT    256
#define EDIM  300
#define EPAD  320
#define HD    256
#define G4    1024     // 4*HD gate width
#define KC_N  18       // (EPAD+HD)/32 k-chunks
#define NT_N  64       // G4/16 n-tiles
#define KTAG  20
#define KPAD  32
#define TAG_START 18
#define TAG_STOP  19
#define NEGV  (-10000.0f)

// ---------------------------------------------------------------------------
// Kernel 1: embedding gather -> bf16, time-major rows [T*B, EPAD], zero pad.
// ---------------------------------------------------------------------------
__global__ void embed_gather(const int* __restrict__ sent,
                             const float* __restrict__ table,
                             bf16* __restrict__ xbf) {
  const long total = (long)TT * BN * EPAD;
  for (long idx = (long)blockIdx.x * blockDim.x + threadIdx.x; idx < total;
       idx += (long)gridDim.x * blockDim.x) {
    long row = idx / EPAD;
    int  e   = (int)(idx - row * EPAD);
    int  t   = (int)(row / BN);
    int  b   = (int)(row - (long)t * BN);
    float v = 0.0f;
    if (e < EDIM) {
      int s = sent[b * TT + t];           // sentence is [B, T] row-major
      v = table[(long)s * EDIM + e];
    }
    xbf[idx] = (bf16)v;
  }
}

// ---------------------------------------------------------------------------
// Kernel 2: build pre-swizzled bf16 WMMA B-fragments.
// B 32x16 bf16 layout (wave32): lane = h*16 + n%16 ; element e=0..15 of
// the lane's 32B covers k = kc*32 + (e/8)*16 + (lane/16)*8 + e%8, n = nt*16 + lane%16.
// ---------------------------------------------------------------------------
#define DIR_ELEMS (KC_N * NT_N * 512)     // 589824 bf16 per direction
#define WO_ELEMS  (16 * 2 * 512)          // 16384 bf16

__global__ void prep_weights(const float* __restrict__ wihf, const float* __restrict__ whhf,
                             const float* __restrict__ wihb, const float* __restrict__ whhb,
                             const float* __restrict__ wout,
                             bf16* __restrict__ wf, bf16* __restrict__ wb,
                             bf16* __restrict__ wo) {
  const long total = 2L * DIR_ELEMS + WO_ELEMS;
  for (long idx = (long)blockIdx.x * blockDim.x + threadIdx.x; idx < total;
       idx += (long)gridDim.x * blockDim.x) {
    if (idx < 2L * DIR_ELEMS) {
      int d = (idx >= DIR_ELEMS) ? 1 : 0;
      int i = (int)(idx - (long)d * DIR_ELEMS);
      int fragid = i >> 9;                 // /512
      int r      = i & 511;
      int lane   = r >> 4;
      int e      = r & 15;
      int kc = fragid >> 6;                // /NT_N
      int nt = fragid & 63;
      int k = kc * 32 + (e >> 3) * 16 + (lane >> 4) * 8 + (e & 7);
      int n = nt * 16 + (lane & 15);
      const float* wih = d ? wihb : wihf;
      const float* whh = d ? whhb : whhf;
      float v = 0.0f;
      if (k < EDIM)            v = wih[n * EDIM + k];
      else if (k >= EPAD)      v = whh[n * HD + (k - EPAD)];
      (d ? wb : wf)[i] = (bf16)v;
    } else {
      int i = (int)(idx - 2L * DIR_ELEMS);
      int fragid = i >> 9;
      int r      = i & 511;
      int lane   = r >> 4;
      int e      = r & 15;
      int kc = fragid >> 1;
      int nt = fragid & 1;
      int k = kc * 32 + (e >> 3) * 16 + (lane >> 4) * 8 + (e & 7);
      int n = nt * 16 + (lane & 15);
      float v = (n < KTAG) ? wout[n * 512 + k] : 0.0f;
      wo[i] = (bf16)v;
    }
  }
}

// ---------------------------------------------------------------------------
// Kernel 3: persistent BiLSTM. grid = {dir 0 fwd, dir 1 bwd}, 1024 threads.
// Wave w owns batch-tile m = w>>3 and hidden cols [ (w&7)*32, +32 ): its 8 C
// tiles are the 4 gates (i,f,g,o) x 2 column-tiles -> wave-local LSTM cell.
// An opaque 32-bit offset (asm) is folded into the weight base each step so
// LICM cannot hoist the 288 fragment loads out of the time loop, while the
// pointer keeps its global address space (global_load_b128, not flat).
// ---------------------------------------------------------------------------
__global__ __launch_bounds__(1024) void lstm_kernel(
    const bf16* __restrict__ xbf,
    const bf16* __restrict__ wfrag_f, const bf16* __restrict__ wfrag_b,
    const float* __restrict__ bias_f, const float* __restrict__ bias_b,
    bf16* __restrict__ hcat) {
  __shared__ bf16 h_prev[BN * HD];       // 32 KB

  const int d = blockIdx.x;
  const bf16*  wfrag = d ? wfrag_b : wfrag_f;
  const float* bias  = d ? bias_b  : bias_f;

  const int tid  = threadIdx.x;
  const int w    = tid >> 5;
  const int lane = tid & 31;
  const int m    = w >> 3;               // 0..3  (16-row batch tile)
  const int ngrp = w & 7;                // owns hidden cols [ngrp*32, +32)
  const int lm   = lane & 15;
  const int lh   = lane >> 4;

  // preload per-lane gate biases (column j fixed per half)
  float bI[2], bF[2], bG[2], bO[2];
  #pragma unroll
  for (int half = 0; half < 2; ++half) {
    const int j = (ngrp * 2 + half) * 16 + lm;
    bI[half] = bias[j];
    bF[half] = bias[HD + j];
    bG[half] = bias[2 * HD + j];
    bO[half] = bias[3 * HD + j];
  }

  for (int i = tid; i < BN * HD; i += 1024) h_prev[i] = (bf16)0.0f;
  float cst[16];
  #pragma unroll
  for (int i = 0; i < 16; ++i) cst[i] = 0.0f;
  __syncthreads();

  const bf16* hrow = h_prev + (m * 16 + lm) * HD + lh * 8;

  for (int step = 0; step < TT; ++step) {
    const int tt = d ? (TT - 1 - step) : step;
    const bf16* xrow = xbf + ((size_t)(tt * BN + m * 16 + lm)) * EPAD + lh * 8;

    // Opaque zero offset: loop-variant weight addresses (defeats LICM) while
    // preserving the global address space of the base pointer.
    int zoff = 0;
    asm volatile("" : "+v"(zoff));
    const bf16* wstep = wfrag + zoff;

    // prefetch next timestep's x row while we compute this one
    if (step + 1 < TT) {
      const int ttn = d ? (tt - 1) : (tt + 1);
      __builtin_prefetch(xbf + ((size_t)(ttn * BN + m * 16 + lm)) * EPAD + lh * 8, 0, 1);
    }

    // ---------------- Phase 1: WMMA gate GEMM ----------------
    v8f C[8];
    #pragma unroll
    for (int s = 0; s < 8; ++s)
      #pragma unroll
      for (int r = 0; r < 8; ++r) C[s][r] = 0.0f;

    #pragma unroll
    for (int kc = 0; kc < KC_N; ++kc) {
      FragBF A;
      if (kc < EPAD / 32) {              // x_t part from global (compile-time)
        A.q[0] = *(const f32x4*)(xrow + kc * 32);
        A.q[1] = *(const f32x4*)(xrow + kc * 32 + 16);
      } else {                           // h_{t-1} part from LDS
        A.q[0] = *(const f32x4*)(hrow + (kc - EPAD / 32) * 32);
        A.q[1] = *(const f32x4*)(hrow + (kc - EPAD / 32) * 32 + 16);
      }
      #pragma unroll
      for (int s = 0; s < 8; ++s) {
        // s = half*4 + gate: nt = gate*16 + (ngrp*2 + half)
        const int nt = (s & 3) * 16 + ngrp * 2 + (s >> 2);
        const bf16* bsrc = wstep + ((size_t)(kc * NT_N + nt) * 512) + lane * 16;
        FragBF Bf;
        Bf.q[0] = *(const f32x4*)(bsrc);
        Bf.q[1] = *(const f32x4*)(bsrc + 8);
        C[s] = __builtin_amdgcn_wmma_f32_16x16x32_bf16(
            false, A.v, false, Bf.v, (short)0, C[s], false, false);
      }
    }
    __syncthreads();   // all h_prev reads done before anyone overwrites it

    // ---------------- Phase 2: wave-local LSTM cell ----------------
    #pragma unroll
    for (int half = 0; half < 2; ++half) {
      const int j = (ngrp * 2 + half) * 16 + lm;   // hidden column
      const v8f gi = C[half * 4 + 0];
      const v8f gf = C[half * 4 + 1];
      const v8f gg = C[half * 4 + 2];
      const v8f go = C[half * 4 + 3];
      #pragma unroll
      for (int r = 0; r < 8; ++r) {
        const int b = m * 16 + lh * 8 + r;         // batch row
        float si = 1.0f / (1.0f + __expf(-(gi[r] + bI[half])));
        float sf = 1.0f / (1.0f + __expf(-(gf[r] + bF[half])));
        float so = 1.0f / (1.0f + __expf(-(go[r] + bO[half])));
        float c  = sf * cst[half * 8 + r] + si * tanhf(gg[r] + bG[half]);
        cst[half * 8 + r] = c;
        float h = so * tanhf(c);
        bf16 h16 = (bf16)h;
        h_prev[b * HD + j] = h16;
        hcat[((size_t)tt * BN + b) * (2 * HD) + d * HD + j] = h16;
      }
    }
    __syncthreads();   // h_prev fully updated before next step reads it
  }
}

// ---------------------------------------------------------------------------
// Kernel 4: feats = hcat @ W_out^T + b_out  (WMMA, N padded to 32).
// ---------------------------------------------------------------------------
__global__ __launch_bounds__(256) void feats_kernel(
    const bf16* __restrict__ hcat, const bf16* __restrict__ wo,
    const float* __restrict__ bout, float* __restrict__ feats) {
  const int tid  = threadIdx.x;
  const int lane = tid & 31;
  const int gw   = blockIdx.x * 8 + (tid >> 5);   // 0..1023 row-tiles
  const int lm   = lane & 15;
  const int lh   = lane >> 4;

  v8f C[2];
  #pragma unroll
  for (int s = 0; s < 2; ++s)
    #pragma unroll
    for (int r = 0; r < 8; ++r) C[s][r] = 0.0f;

  #pragma unroll
  for (int kc = 0; kc < 16; ++kc) {
    FragBF A;
    const bf16* arow = hcat + ((size_t)(gw * 16 + lm)) * 512 + kc * 32 + lh * 8;
    A.q[0] = *(const f32x4*)(arow);
    A.q[1] = *(const f32x4*)(arow + 16);
    #pragma unroll
    for (int nt = 0; nt < 2; ++nt) {
      const bf16* bsrc = wo + ((size_t)(kc * 2 + nt) * 512) + lane * 16;
      FragBF Bf;
      Bf.q[0] = *(const f32x4*)(bsrc);
      Bf.q[1] = *(const f32x4*)(bsrc + 8);
      C[nt] = __builtin_amdgcn_wmma_f32_16x16x32_bf16(
          false, A.v, false, Bf.v, (short)0, C[nt], false, false);
    }
  }
  #pragma unroll
  for (int nt = 0; nt < 2; ++nt) {
    const int col = nt * 16 + lm;
    const float bo = (col < KTAG) ? bout[col] : 0.0f;
    #pragma unroll
    for (int r = 0; r < 8; ++r) {
      const int row = gw * 16 + lh * 8 + r;
      feats[(size_t)row * KPAD + col] = C[nt][r] + bo;
    }
  }
}

// ---------------------------------------------------------------------------
// Kernel 5: Viterbi. One 32-thread block per batch element; lane = next tag.
// ---------------------------------------------------------------------------
__global__ __launch_bounds__(32) void viterbi_kernel(
    const float* __restrict__ feats, const int* __restrict__ length,
    const float* __restrict__ trans, int* __restrict__ bp,
    float* __restrict__ out) {
  const int b = blockIdx.x;
  const int k = threadIdx.x;
  __shared__ float fv[KTAG];
  __shared__ float ts[KTAG * KTAG];
  __shared__ float red_s[KTAG];

  for (int i = k; i < KTAG * KTAG; i += 32) ts[i] = trans[i];
  if (k < KTAG) fv[k] = (k == TAG_START) ? 0.0f : NEGV;
  __syncthreads();

  const int len = length[b];
  for (int t = 0; t < TT; ++t) {
    float fvn = 0.0f; int bpv = k;
    if (k < KTAG) {
      float best = -3.4e38f; int arg = 0;
      #pragma unroll
      for (int p = 0; p < KTAG; ++p) {
        float sc = fv[p] + ts[k * KTAG + p];
        if (sc > best) { best = sc; arg = p; }
      }
      float feat = feats[((size_t)t * BN + b) * KPAD + k];
      bool valid = (t < len);
      fvn = valid ? (best + feat) : fv[k];
      bpv = valid ? arg : k;
    }
    __syncthreads();
    if (k < KTAG) {
      fv[k] = fvn;
      bp[((size_t)t * BN + b) * KTAG + k] = bpv;
    }
    __syncthreads();
  }

  if (k < KTAG) red_s[k] = fv[k] + ts[TAG_STOP * KTAG + k];
  __syncthreads();
  if (k == 0) {
    float best = red_s[0]; int arg = 0;
    for (int p = 1; p < KTAG; ++p)
      if (red_s[p] > best) { best = red_s[p]; arg = p; }
    out[b] = best;                        // viterbi score
    int tag = arg;
    for (int t = TT - 1; t >= 0; --t) {
      out[BN + b * TT + t] = (float)tag;  // path, time order
      tag = bp[((size_t)t * BN + b) * KTAG + tag];
    }
  }
}

// ---------------------------------------------------------------------------
// Host launcher
// ---------------------------------------------------------------------------
extern "C" void kernel_launch(void* const* d_in, const int* in_sizes, int n_in,
                              void* d_out, int out_size, void* d_ws, size_t ws_size,
                              hipStream_t stream) {
  const int*   sent   = (const int*)d_in[0];
  const int*   length = (const int*)d_in[1];
  const float* table  = (const float*)d_in[2];
  const float* wihf   = (const float*)d_in[3];
  const float* whhf   = (const float*)d_in[4];
  const float* bf_    = (const float*)d_in[5];
  const float* wihb   = (const float*)d_in[6];
  const float* whhb   = (const float*)d_in[7];
  const float* bb_    = (const float*)d_in[8];
  const float* wout   = (const float*)d_in[9];
  const float* bout   = (const float*)d_in[10];
  const float* trans  = (const float*)d_in[11];

  char* ws = (char*)d_ws;
  // workspace layout (256B aligned), total ~33.1 MB
  bf16*  xbf   = (bf16*)(ws + 0);            // 16384*320 bf16  = 10,485,760 B
  bf16*  wf    = (bf16*)(ws + 10485760);     // 589,824 bf16    =  1,179,648 B
  bf16*  wb    = (bf16*)(ws + 11665408);     // 589,824 bf16    =  1,179,648 B
  bf16*  wo    = (bf16*)(ws + 12845056);     // 16,384 bf16     =     32,768 B
  bf16*  hcat  = (bf16*)(ws + 12877824);     // 16384*512 bf16  = 16,777,216 B
  float* feats = (float*)(ws + 29655040);    // 16384*32 f32    =  2,097,152 B
  int*   bp    = (int*)(ws + 31752192);      // 256*64*20 i32   =  1,310,720 B

  embed_gather<<<2048, 256, 0, stream>>>(sent, table, xbf);
  prep_weights<<<1200, 256, 0, stream>>>(wihf, whhf, wihb, whhb, wout, wf, wb, wo);
  lstm_kernel<<<2, 1024, 0, stream>>>(xbf, wf, wb, bf_, bb_, hcat);
  feats_kernel<<<128, 256, 0, stream>>>(hcat, wo, bout, feats);
  viterbi_kernel<<<64, 32, 0, stream>>>(feats, length, trans, bp, (float*)d_out);
}
